// Conv2d_NN_sanity_23338852287089
// MI455X (gfx1250) — matrix operand order repaired and checked by reading, not compiled, not run on hardware
//
#include <hip/hip_runtime.h>
#include <hip/hip_bf16.h>

// ---------------------------------------------------------------------------
// Conv2d_NN: cosine-sim kNN (K=9) + conv1d(kernel=9, stride=9) for gfx1250.
//   x: [8, 32, 56, 56] f32     W: [64, 32, 9] f32     out: [8, 64, 56, 56] f32
// Pipeline:
//   1) normalize rows -> xn f16, t-major [b][t][c]                 (VALU)
//   2) sim = Xn^T Xn via v_wmma_f32_16x16x32_f16 (K == C == 32),
//      2 t-tiles per wave (B operand reused by 2 WMMAs), fused with
//      per-row top-9 selection (1 row per lane, all 32 lanes busy)
//   3) gather + conv1d as 9 chained K=32 WMMAs per (t-tile, o-tile)
// ---------------------------------------------------------------------------

#define BB   8
#define CC   32
#define HH   56
#define TT   (HH * HH)        // 3136
#define KNN  9
#define OC   64
#define NT   (TT / 16)        // 196 tiles of 16 along t
#define NT2  (TT / 32)        // 98 double-tiles for the sim kernel

typedef __attribute__((ext_vector_type(16))) _Float16 v16h;
typedef __attribute__((ext_vector_type(8)))  float    v8f;

// ---------------------------------------------------------------- kernel 1 --
// Per (b,t): L2-normalize across C, store as f16 t-major: xn[(b*T+t)*32 + c].
__global__ void nn_normalize(const float* __restrict__ x,
                             _Float16* __restrict__ xn) {
    int tid = blockIdx.x * blockDim.x + threadIdx.x;
    if (tid >= BB * TT) return;
    int b = tid / TT, t = tid - b * TT;
    const float* xb = x + (size_t)b * CC * TT + t;
    float v[CC];
    float s = 0.f;
#pragma unroll
    for (int c = 0; c < CC; ++c) { v[c] = xb[c * TT]; s += v[c] * v[c]; }
    float inv = 1.f / fmaxf(sqrtf(s), 1e-12f);
    _Float16* o = xn + (size_t)tid * CC;
#pragma unroll
    for (int c = 0; c < CC; ++c) o[c] = (_Float16)(v[c] * inv);
}

// ---------------------------------------------------------------- kernel 1b -
// Re-layout conv weights to f16: Wh[k][o][c]  (c contiguous for WMMA B loads).
__global__ void nn_prep_w(const float* __restrict__ W,
                          _Float16* __restrict__ Wh) {
    int tid = blockIdx.x * blockDim.x + threadIdx.x;
    if (tid >= KNN * OC * CC) return;
    int c = tid & (CC - 1);
    int o = (tid / CC) % OC;
    int k = tid / (CC * OC);
    Wh[tid] = (_Float16)W[(o * CC + c) * KNN + k];
}

// ---------------------------------------------------------------- kernel 2 --
// One wave per (double-t-tile, b): 32 sim rows per wave. Loop over 196
// s-tiles: one B load -> two WMMAs (rows t0..t0+15 and t0+16..t0+31),
// spill 32x16 slab to LDS, each lane owns exactly one row's top-9.
// Next-iteration B operand is prefetched before the selection phase so the
// global loads overlap the VALU top-K work.
__global__ void __launch_bounds__(32)
nn_sim_topk(const _Float16* __restrict__ xn, int* __restrict__ idx) {
    __shared__ float tile[32][17];           // +1 pad: conflict-free row reads

    const int lane = threadIdx.x;
    const int l16  = lane & 15;
    const int hi   = lane >> 4;              // 0: lanes 0-15, 1: lanes 16-31
    const int t0   = blockIdx.x * 32;
    const int b    = blockIdx.y;

    const _Float16* base = xn + (size_t)b * TT * CC;
    const int cb = hi * 16;

    // ---- A operands: rows t0..t0+15 (a0) and t0+16..t0+31 (a1), resident.
    // lane<16 : M=l16, K = 0..15;  lane>=16: M=l16, K = 16..31
    v16h a0, a1;
    {
        const _Float16* r0 = base + (size_t)(t0 + l16) * CC;
        const _Float16* r1 = base + (size_t)(t0 + 16 + l16) * CC;
#pragma unroll
        for (int i = 0; i < 8; ++i) {
            a0[i]     = r0[cb + i];
            a0[8 + i] = r0[cb + 8 + i];
            a1[i]     = r1[cb + i];
            a1[8 + i] = r1[cb + 8 + i];
        }
    }

    // ---- top-9 state (ascending by value); lane owns row t = t0 + lane.
    float bv[KNN];
    int   bi[KNN];
#pragma unroll
    for (int p = 0; p < KNN; ++p) { bv[p] = -2.0f; bi[p] = -1; }
    const int t = t0 + lane;

    // ---- prefetch B(0): lane<16: N=l16, K=0..15 ; lane>=16: N=l16, K=16..31
    v16h bm;
    {
        const _Float16* bc = base + (size_t)l16 * CC + hi * 16;
#pragma unroll
        for (int i = 0; i < 16; ++i) bm[i] = bc[i];
    }

    for (int j = 0; j < NT; ++j) {
        v8f c0 = {}, c1 = {};
        c0 = __builtin_amdgcn_wmma_f32_16x16x32_f16(
                 false, a0, false, bm, (short)0, c0, false, false);
        c1 = __builtin_amdgcn_wmma_f32_16x16x32_f16(
                 false, a1, false, bm, (short)0, c1, false, false);

        // prefetch next B while we do the selection below
        v16h bmn;
        {
            const int jn = (j + 1 < NT) ? j + 1 : 0;   // wrapped, stays in-bounds
            const _Float16* bc = base + (size_t)(jn * 16 + l16) * CC + hi * 16;
#pragma unroll
            for (int i = 0; i < 16; ++i) bmn[i] = bc[i];
        }

        // D: lane holds D[r + 8*hi][l16], r = 0..7  -> spill 32x16 slab
#pragma unroll
        for (int r = 0; r < 8; ++r) {
            tile[r + 8 * hi][l16]      = c0[r];
            tile[16 + r + 8 * hi][l16] = c1[r];
        }
        __syncthreads();

        const int s0 = j * 16;
#pragma unroll 4
        for (int n = 0; n < 16; ++n) {
            float v = tile[lane][n];
            v = fminf(1.0f, fmaxf(-1.0f, v));          // clip
            const int s = s0 + n;
            if (s == t) v = 1.1f;                      // self always top-1
            if (v > bv[0]) {                           // evict current min
#pragma unroll
                for (int p = 0; p < KNN; ++p) {
                    if (p < KNN - 1 && v > bv[p + 1]) {
                        bv[p] = bv[p + 1]; bi[p] = bi[p + 1];
                    } else {
                        bv[p] = v; bi[p] = s;
                        break;
                    }
                }
            }
        }
        __syncthreads();
        bm = bmn;
    }

    // sort the 9 winning indices ascending (matches jnp.sort(idx))
#pragma unroll
    for (int i = 0; i < KNN - 1; ++i)
#pragma unroll
        for (int jj = 0; jj < KNN - 1 - i; ++jj)
            if (bi[jj] > bi[jj + 1]) {
                int tmp = bi[jj]; bi[jj] = bi[jj + 1]; bi[jj + 1] = tmp;
            }
    int* op = idx + ((size_t)b * TT + t) * KNN;
#pragma unroll
    for (int p = 0; p < KNN; ++p) op[p] = bi[p];
}

// ---------------------------------------------------------------- kernel 3 --
// One wave per (t-tile, b): out[16t x 64o] = Gather(16t x 288) x W(288 x 64o)
// as 9 chained K=32 WMMAs per o-tile (4 o-tiles, accumulators in VGPRs).
__global__ void __launch_bounds__(32)
nn_gather_conv(const float* __restrict__ x, const _Float16* __restrict__ Wh,
               const int* __restrict__ idx, float* __restrict__ out) {
    __shared__ int sidx[16][KNN];

    const int lane = threadIdx.x;
    const int l16  = lane & 15;
    const int hi   = lane >> 4;
    const int t0   = blockIdx.x * 16;
    const int b    = blockIdx.y;

    if (lane < 16) {
        const int* ip = idx + ((size_t)b * TT + t0 + lane) * KNN;
#pragma unroll
        for (int k = 0; k < KNN; ++k) sidx[lane][k] = ip[k];
    }
    __syncthreads();

    const float* xb = x + (size_t)b * CC * TT;
    v8f acc[4] = {{}, {}, {}, {}};
    const int cb = hi * 16;  // A-operand K range: cb..cb+15

    for (int k = 0; k < KNN; ++k) {
        // A: A[m][c] = x[b][c][ idx[t0+m][k] ], m = l16 (gather + f32->f16)
        const int t_idx = sidx[l16][k];
        v16h a;
#pragma unroll
        for (int i = 0; i < 8; ++i)
            a[i]     = (_Float16)xb[(cb + i) * TT + t_idx];
#pragma unroll
        for (int i = 0; i < 8; ++i)
            a[8 + i] = (_Float16)xb[(cb + 8 + i) * TT + t_idx];

#pragma unroll
        for (int ot = 0; ot < 4; ++ot) {
            // B: B[c][o] = Wh[k][o][c], column o = ot*16 + l16,
            //    lane<16: c=0..15 ; lane>=16: c=16..31   (contiguous halves)
            const _Float16* wp =
                Wh + ((size_t)(k * OC) + ot * 16 + l16) * CC + hi * 16;
            v16h bm;
#pragma unroll
            for (int i = 0; i < 16; ++i) bm[i] = wp[i];
            acc[ot] = __builtin_amdgcn_wmma_f32_16x16x32_f16(
                          false, a, false, bm, (short)0, acc[ot], false, false);
        }
    }

    // D: lane holds rows t = t0 + 8*hi + r (contiguous), col o = o0 + l16
#pragma unroll
    for (int ot = 0; ot < 4; ++ot) {
        const int o = ot * 16 + l16;
        float* op = out + ((size_t)b * OC + o) * TT + t0 + 8 * hi;
#pragma unroll
        for (int r = 0; r < 8; ++r) op[r] = acc[ot][r];
    }
}

// ---------------------------------------------------------------------------
extern "C" void kernel_launch(void* const* d_in, const int* in_sizes, int n_in,
                              void* d_out, int out_size, void* d_ws, size_t ws_size,
                              hipStream_t stream) {
    const float* x = (const float*)d_in[0];   // [8,32,56,56]
    const float* W = (const float*)d_in[1];   // [64,32,9]
    float* out = (float*)d_out;               // [8,64,56,56]

    char* ws = (char*)d_ws;
    const size_t XN_BYTES = (size_t)BB * TT * CC * sizeof(_Float16); // 1,605,632
    const size_t WH_BYTES = (size_t)KNN * OC * CC * sizeof(_Float16); //   36,864
    _Float16* xn  = (_Float16*)ws;
    _Float16* Wh  = (_Float16*)(ws + XN_BYTES);
    int*      idx = (int*)(ws + XN_BYTES + WH_BYTES);                //  903,168

    nn_normalize<<<(BB * TT + 255) / 256, 256, 0, stream>>>(x, xn);
    nn_prep_w<<<(KNN * OC * CC + 255) / 256, 256, 0, stream>>>(W, Wh);
    nn_sim_topk<<<dim3(NT2, BB), 32, 0, stream>>>(xn, idx);
    nn_gather_conv<<<dim3(NT, BB), 32, 0, stream>>>(x, Wh, idx, out);
}
